// MemoryEfficientAttentionVarlen_40226663694471
// MI455X (gfx1250) — matrix operand order repaired and checked by reading, compiled, not run on hardware
//
#include <hip/hip_runtime.h>

typedef __attribute__((ext_vector_type(16))) __bf16 v16bf;
typedef __attribute__((ext_vector_type(8)))  float  v8f;
typedef unsigned short ushort_t;
typedef unsigned int   uint_t;

union FragAB { uint4 q[2]; v16bf v; };   // 32 bytes: 16 bf16 (8 VGPRs)
union FragC  { float f[8]; v8f  v; };    // 32 bytes: 8 f32 accumulators

__device__ __forceinline__ ushort_t f2bf(float f) {
    uint_t u = __float_as_uint(f);
    u += 0x7FFFu + ((u >> 16) & 1u);     // round-to-nearest-even
    return (ushort_t)(u >> 16);
}
__device__ __forceinline__ float bf2f(ushort_t h) {
    return __uint_as_float(((uint_t)h) << 16);
}

// Pack two f32 -> packed bf16x2 (v_cvt_pk_bf16_f32 when available)
#if __has_builtin(__builtin_amdgcn_cvt_pk_bf16_f32)
__device__ __forceinline__ uint_t pack2(float a, float b) {
    auto p = __builtin_amdgcn_cvt_pk_bf16_f32(a, b);
    return __builtin_bit_cast(uint_t, p);
}
#else
__device__ __forceinline__ uint_t pack2(float a, float b) {
    return (uint_t)f2bf(a) | ((uint_t)f2bf(b) << 16);
}
#endif

// 16-lane butterfly shuffle: DPP16 ROW_XMASK (VALU, no LDS traffic) w/ fallback
template<int M>
__device__ __forceinline__ float swz(float x) {
#if __has_builtin(__builtin_amdgcn_update_dpp)
    int v = __builtin_amdgcn_update_dpp(0, __float_as_int(x), 0x160 | M, 0xf, 0xf, true);
    return __int_as_float(v);
#else
    return __shfl_xor(x, M, 32);
#endif
}

// CDNA5 async global->LDS copy (ASYNCcnt) via inline asm (portable per ISA doc).
// VDST VGPR = LDS byte address (low 32 bits of generic pointer), GV mode.
__device__ __forceinline__ void async_load_b128(void* lds_dst, const void* gsrc) {
    asm volatile("global_load_async_to_lds_b128 %0, %1, off"
                 :: "v"((uint_t)(uintptr_t)lds_dst),
                    "v"((unsigned long long)(uintptr_t)gsrc)
                 : "memory");
}
__device__ __forceinline__ void wait_async() {
    asm volatile("s_wait_asynccnt 0x0" ::: "memory");
}

#define WMMA_BF16(A, B, C) \
    __builtin_amdgcn_wmma_f32_16x16x32_bf16(false, (A), false, (B), (short)0, (C), false, false)

// ---------------------------------------------------------------------------
// Tiled GEMM: C[M,N] = A[M,K] * B[N,K]^T    (B given row-major as (N,K))
// Block tile 128x128x32, 256 threads = 8 waves, wave tile 64x32 (4x2 WMMA).
// ---------------------------------------------------------------------------
template<bool A_BF16, bool OUT_F32>
__global__ __launch_bounds__(256) void gemm_bt_kernel(
    const void* __restrict__ Ap, const float* __restrict__ B,
    void* __restrict__ Cp, int M, int N, int K, int lda, int ldb, int ldc)
{
    __shared__ ushort_t As[128][40];   // 32 K + 8 pad (16B-aligned rows, spread banks)
    __shared__ ushort_t Bs[128][40];

    const int t    = threadIdx.x;
    const int lane = t & 31;
    const int wid  = t >> 5;
    const int waveM = (wid >> 2) * 64;
    const int waveN = (wid & 3) * 32;
    const int m0 = blockIdx.y * 128;
    const int n0 = blockIdx.x * 128;
    const int hi = lane >> 4;
    const int cl = lane & 15;

    FragC acc[4][2];
    #pragma unroll
    for (int i = 0; i < 4; ++i)
        #pragma unroll
        for (int j = 0; j < 2; ++j)
            #pragma unroll
            for (int r = 0; r < 8; ++r) acc[i][j].f[r] = 0.f;

    const int rowL = t >> 1;           // 128 rows, 2 threads per row
    const int colL = (t & 1) * 16;     // 16 K-elements each

    for (int k0 = 0; k0 < K; k0 += 32) {
        // ---- stage A tile into LDS ----
        if constexpr (A_BF16) {
            // data already bf16: CDNA5 async DMA straight into LDS
            const ushort_t* src = (const ushort_t*)Ap + (size_t)(m0 + rowL) * lda + k0 + colL;
            async_load_b128(&As[rowL][colL],     src);
            async_load_b128(&As[rowL][colL + 8], src + 8);
        } else {
            const float* src = (const float*)Ap + (size_t)(m0 + rowL) * lda + k0 + colL;
            float4 f0 = ((const float4*)src)[0];
            float4 f1 = ((const float4*)src)[1];
            float4 f2 = ((const float4*)src)[2];
            float4 f3 = ((const float4*)src)[3];
            uint4 p0, p1;
            p0.x = pack2(f0.x, f0.y); p0.y = pack2(f0.z, f0.w);
            p0.z = pack2(f1.x, f1.y); p0.w = pack2(f1.z, f1.w);
            p1.x = pack2(f2.x, f2.y); p1.y = pack2(f2.z, f2.w);
            p1.z = pack2(f3.x, f3.y); p1.w = pack2(f3.z, f3.w);
            *(uint4*)&As[rowL][colL]     = p0;
            *(uint4*)&As[rowL][colL + 8] = p1;
        }
        // ---- stage B tile (always f32 in, convert in flight) ----
        {
            const float* src = B + (size_t)(n0 + rowL) * ldb + k0 + colL;
            float4 f0 = ((const float4*)src)[0];
            float4 f1 = ((const float4*)src)[1];
            float4 f2 = ((const float4*)src)[2];
            float4 f3 = ((const float4*)src)[3];
            uint4 p0, p1;
            p0.x = pack2(f0.x, f0.y); p0.y = pack2(f0.z, f0.w);
            p0.z = pack2(f1.x, f1.y); p0.w = pack2(f1.z, f1.w);
            p1.x = pack2(f2.x, f2.y); p1.y = pack2(f2.z, f2.w);
            p1.z = pack2(f3.x, f3.y); p1.w = pack2(f3.z, f3.w);
            *(uint4*)&Bs[rowL][colL]     = p0;
            *(uint4*)&Bs[rowL][colL + 8] = p1;
        }
        if constexpr (A_BF16) wait_async();
        __syncthreads();

        // ---- fragments per ISA 16-bit layouts ----
        const int klo  = hi * 8;    // A: two 8-elem chunks at klo and klo+16
        const int klo2 = hi * 8;    // B: two 8-elem chunks at klo2 and klo2+16
        FragAB af[4], bfr[2];
        #pragma unroll
        for (int mt = 0; mt < 4; ++mt) {
            int row = waveM + mt * 16 + cl;
            af[mt].q[0] = *(const uint4*)&As[row][klo];
            af[mt].q[1] = *(const uint4*)&As[row][klo + 16];
        }
        #pragma unroll
        for (int nt = 0; nt < 2; ++nt) {
            int row = waveN + nt * 16 + cl;
            bfr[nt].q[0] = *(const uint4*)&Bs[row][klo2];
            bfr[nt].q[1] = *(const uint4*)&Bs[row][klo2 + 16];
        }
        #pragma unroll
        for (int mt = 0; mt < 4; ++mt)
            #pragma unroll
            for (int nt = 0; nt < 2; ++nt)
                acc[mt][nt].v = WMMA_BF16(af[mt].v, bfr[nt].v, acc[mt][nt].v);
        __syncthreads();
    }

    // ---- store: C layout lane cl = col, rows r + 8*hi ----
    #pragma unroll
    for (int mt = 0; mt < 4; ++mt)
        #pragma unroll
        for (int nt = 0; nt < 2; ++nt)
            #pragma unroll
            for (int r = 0; r < 8; ++r) {
                int row = m0 + waveM + mt * 16 + r + 8 * hi;
                int col = n0 + waveN + nt * 16 + cl;
                if constexpr (OUT_F32)
                    ((float*)Cp)[(size_t)row * ldc + col] = acc[mt][nt].f[r];
                else
                    ((ushort_t*)Cp)[(size_t)row * ldc + col] = f2bf(acc[mt][nt].f[r]);
            }
}

// ---------------------------------------------------------------------------
// RoPE on q and k halves of qkv (bf16, layout [s][12288]); folds 1/sqrt(128)
// into q. Each thread owns the (d, d+64) pair for one (which,s,head).
// ---------------------------------------------------------------------------
__global__ __launch_bounds__(256) void rope_kernel(ushort_t* __restrict__ qkvbf,
                                                   const long long* __restrict__ pos)
{
    int idx   = blockIdx.x * 256 + threadIdx.x;     // 2*2048*32*64 = 2^23 threads
    int d     = idx & 63;
    int head  = (idx >> 6) & 31;
    int s     = (idx >> 11) & 2047;
    int which = idx >> 22;                          // 0 = q, 1 = k
    size_t base = (size_t)s * 12288 + (size_t)which * 4096 + head * 128 + d;

    float p    = (float)pos[s];
    float invf = __expf(-(float)d * 0.14391156831212787f); // ln(10000)/64
    float f    = p * invf;
    float sn, c;
    __sincosf(f, &sn, &c);

    float x0 = bf2f(qkvbf[base]);
    float x1 = bf2f(qkvbf[base + 64]);
    float o0 = x0 * c - x1 * sn;
    float o1 = x1 * c + x0 * sn;
    if (which == 0) { o0 *= 0.08838834764831845f; o1 *= 0.08838834764831845f; }
    qkvbf[base]      = f2bf(o0);
    qkvbf[base + 64] = f2bf(o1);
}

// ---------------------------------------------------------------------------
// Flash attention: one block per (128-query tile, head). 8 waves, wave owns
// 16 query rows. Q frags in registers; K tile async-DMAed into LDS; V tile
// transposed through registers; online softmax with DPP16 row butterflies;
// P staged through wave-private LDS for the P*V WMMAs.
// ---------------------------------------------------------------------------
__global__ __launch_bounds__(256) void attn_kernel(const ushort_t* __restrict__ qkvbf,
                                                   ushort_t* __restrict__ attnbf)
{
    __shared__ ushort_t Ks[64][136];   // [kv][d]   (B for S = Q K^T)
    __shared__ ushort_t Vs[128][72];   // [d][kv]   (B for O = P V)
    __shared__ ushort_t Ps[128][72];   // [q][kv]   (A for O = P V)

    const int t    = threadIdx.x;
    const int lane = t & 31;
    const int wid  = t >> 5;
    const int head = blockIdx.y;
    const int q0   = blockIdx.x * 128;
    const int qr   = wid * 16;
    const int hi   = lane >> 4;
    const int cl   = lane & 15;

    const ushort_t* Qg = qkvbf + (size_t)head * 128;
    const ushort_t* Kg = qkvbf + 4096 + (size_t)head * 128;
    const ushort_t* Vg = qkvbf + 8192 + (size_t)head * 128;

    // Q fragments (rows qr..qr+15, K = d in 4 chunks of 32) held in registers
    FragAB qf[4];
    {
        const ushort_t* qrow = Qg + (size_t)(q0 + qr + cl) * 12288;
        #pragma unroll
        for (int kc = 0; kc < 4; ++kc) {
            int k = hi * 8 + kc * 32;
            qf[kc].q[0] = *(const uint4*)(qrow + k);
            qf[kc].q[1] = *(const uint4*)(qrow + k + 16);
        }
    }

    FragC o[8];
    #pragma unroll
    for (int ot = 0; ot < 8; ++ot)
        #pragma unroll
        for (int r = 0; r < 8; ++r) o[ot].f[r] = 0.f;
    float m_i[8], l_i[8];
    #pragma unroll
    for (int r = 0; r < 8; ++r) { m_i[r] = -1e30f; l_i[r] = 0.f; }

    const int nkv = (q0 + 128) / 64;   // causal: only tiles up to the diagonal
    for (int kt = 0; kt < nkv; ++kt) {
        const int k0 = kt * 64;

        // K tile [64][128]: async DMA global -> LDS (8-elem chunks)
        #pragma unroll
        for (int i = 0; i < 4; ++i) {
            int cid = t + i * 256;
            int row = cid >> 4;
            int col = (cid & 15) * 8;
            async_load_b128(&Ks[row][col], Kg + (size_t)(k0 + row) * 12288 + col);
        }
        // V tile transposed -> Vs[d][kv] (needs register pass)
        #pragma unroll
        for (int i = 0; i < 4; ++i) {
            int cid = t + i * 256;
            int row = cid >> 4;
            int col = (cid & 15) * 8;
            uint4 v = *(const uint4*)(Vg + (size_t)(k0 + row) * 12288 + col);
            const ushort_t* pv = (const ushort_t*)&v;
            #pragma unroll
            for (int j = 0; j < 8; ++j) Vs[col + j][row] = pv[j];
        }
        wait_async();
        __syncthreads();

        // ---- S = Q K^T (scale already folded into Q) ----
        FragC s[4];
        #pragma unroll
        for (int nt = 0; nt < 4; ++nt)
            #pragma unroll
            for (int r = 0; r < 8; ++r) s[nt].f[r] = 0.f;
        #pragma unroll
        for (int kc = 0; kc < 4; ++kc) {
            const int klo2 = hi * 8 + kc * 32;
            #pragma unroll
            for (int nt = 0; nt < 4; ++nt) {
                FragAB bk;
                int rowb = nt * 16 + cl;
                bk.q[0] = *(const uint4*)&Ks[rowb][klo2];
                bk.q[1] = *(const uint4*)&Ks[rowb][klo2 + 16];
                s[nt].v = WMMA_BF16(qf[kc].v, bk.v, s[nt].v);
            }
        }

        // ---- causal mask on the diagonal tiles ----
        if (k0 + 63 > q0 + qr) {
            #pragma unroll
            for (int nt = 0; nt < 4; ++nt) {
                int col = k0 + nt * 16 + cl;
                #pragma unroll
                for (int r = 0; r < 8; ++r)
                    if (col > q0 + qr + r + 8 * hi) s[nt].f[r] = -1e30f;
            }
        }

        // ---- online softmax (rows live in 16-lane halves -> DPP ROW_XMASK) ----
        #pragma unroll
        for (int r = 0; r < 8; ++r) {
            float mx = fmaxf(fmaxf(s[0].f[r], s[1].f[r]), fmaxf(s[2].f[r], s[3].f[r]));
            mx = fmaxf(mx, swz<1>(mx));
            mx = fmaxf(mx, swz<2>(mx));
            mx = fmaxf(mx, swz<4>(mx));
            mx = fmaxf(mx, swz<8>(mx));
            float mnew  = fmaxf(m_i[r], mx);
            float alpha = __expf(m_i[r] - mnew);
            float psum  = 0.f;
            #pragma unroll
            for (int nt = 0; nt < 4; ++nt) {
                float pv = __expf(s[nt].f[r] - mnew);
                s[nt].f[r] = pv;
                psum += pv;
            }
            psum += swz<1>(psum);
            psum += swz<2>(psum);
            psum += swz<4>(psum);
            psum += swz<8>(psum);
            l_i[r] = l_i[r] * alpha + psum;
            m_i[r] = mnew;
            #pragma unroll
            for (int ot = 0; ot < 8; ++ot) o[ot].f[r] *= alpha;
        }

        // ---- P -> wave-private LDS region (C layout -> A layout transpose) ----
        #pragma unroll
        for (int nt = 0; nt < 4; ++nt)
            #pragma unroll
            for (int r = 0; r < 8; ++r)
                Ps[qr + r + 8 * hi][nt * 16 + cl] = f2bf(s[nt].f[r]);

        // ---- O += P V ----
        #pragma unroll
        for (int kc = 0; kc < 2; ++kc) {
            FragAB ap;
            int klo = hi * 8 + kc * 32;
            ap.q[0] = *(const uint4*)&Ps[qr + cl][klo];
            ap.q[1] = *(const uint4*)&Ps[qr + cl][klo + 16];
            int klo2 = hi * 8 + kc * 32;
            #pragma unroll
            for (int ot = 0; ot < 8; ++ot) {
                FragAB bv;
                int rowv = ot * 16 + cl;
                bv.q[0] = *(const uint4*)&Vs[rowv][klo2];
                bv.q[1] = *(const uint4*)&Vs[rowv][klo2 + 16];
                o[ot].v = WMMA_BF16(ap.v, bv.v, o[ot].v);
            }
        }
        __syncthreads();   // before next tile overwrites Ks/Vs
    }

    // ---- normalize and emit attn output bf16 in [s][4096] hidden layout ----
    #pragma unroll
    for (int r = 0; r < 8; ++r) {
        float inv = 1.f / l_i[r];
        int rowg  = q0 + qr + r + 8 * hi;
        #pragma unroll
        for (int ot = 0; ot < 8; ++ot) {
            int col = head * 128 + ot * 16 + cl;
            attnbf[(size_t)rowg * 4096 + col] = f2bf(o[ot].f[r] * inv);
        }
    }
}

// ---------------------------------------------------------------------------
extern "C" void kernel_launch(void* const* d_in, const int* in_sizes, int n_in,
                              void* d_out, int out_size, void* d_ws, size_t ws_size,
                              hipStream_t stream)
{
    const float*     hidden  = (const float*)d_in[0];
    const long long* pos     = (const long long*)d_in[1];   // int64 position_ids
    const float*     qkv_mat = (const float*)d_in[2];       // (3,4096,4096) = (12288,4096)
    const float*     o_w     = (const float*)d_in[3];       // (4096,4096)
    float*           out     = (float*)d_out;

    ushort_t* qkvbf  = (ushort_t*)d_ws;                       // 2048 x 12288 bf16 (48 MB)
    ushort_t* attnbf = qkvbf + (size_t)2048 * 12288;          // 2048 x 4096 bf16 (16 MB)

    // 1) QKV projection: (2048x4096) @ (12288x4096)^T -> bf16
    gemm_bt_kernel<false, false><<<dim3(96, 16), 256, 0, stream>>>(
        hidden, qkv_mat, qkvbf, 2048, 12288, 4096, 4096, 4096, 12288);

    // 2) RoPE on q,k (scale folded into q)
    rope_kernel<<<32768, 256, 0, stream>>>(qkvbf, pos);

    // 3) Causal flash attention per (q-tile, head)
    attn_kernel<<<dim3(16, 32), 256, 0, stream>>>(qkvbf, attnbf);

    // 4) Output projection: (2048x4096 bf16) @ (4096x4096)^T -> f32 d_out
    gemm_bt_kernel<true, true><<<dim3(32, 16), 256, 0, stream>>>(
        attnbf, o_w, out, 2048, 4096, 4096, 4096, 4096, 4096);
}
